// BlockTucker_45629732553242
// MI455X (gfx1250) — compile-verified, matrix operand order checked by reading, not run on hardware
//
#include <hip/hip_runtime.h>

#define B_ROWS 8192
#define DIN    2048
#define MMDIM  1600
#define CHUNKS 20
#define CSZ    80
#define KBIL   (CSZ*CSZ)   // 6400
#define DOUT   3000
#define NPAD   3040        // 19 * 160, zero-padded Wout^T rows

#ifndef __has_builtin
#define __has_builtin(x) 0
#endif

typedef __attribute__((ext_vector_type(16))) __bf16          bf16x16;
typedef __attribute__((ext_vector_type(2)))  __bf16          bf16x2;
typedef __attribute__((ext_vector_type(8)))  float           floatx8;
typedef __attribute__((ext_vector_type(4)))  float           f32x4;
typedef __attribute__((ext_vector_type(4)))  unsigned int    u32x4;
typedef __attribute__((ext_vector_type(2)))  unsigned int    u32x2;
typedef __attribute__((ext_vector_type(4)))  unsigned short  u16x4;
typedef __attribute__((ext_vector_type(4)))  int             i32x4;

// address-space qualified pointee types for the async-to-LDS builtin
typedef i32x4 __attribute__((address_space(1))) i32x4_g;
typedef i32x4 __attribute__((address_space(3))) i32x4_l;

union AFrag {
    bf16x16        v;
    u32x4          q[2];
    unsigned int   w[8];
    unsigned short us[16];
};

// f32 -> bf16 via native fptrunc (backend picks best gfx1250 instruction)
__device__ __forceinline__ unsigned short f2bf(float f) {
    union { __bf16 h; unsigned short u; } cv;
    cv.h = (__bf16)f;
    return cv.u;
}
__device__ __forceinline__ float bf2f(unsigned short h) {
    return __uint_as_float(((unsigned int)h) << 16);
}

// packed f32x2 -> bf16x2 (lo = a, hi = b)
#if defined(__AMDGCN__) && __has_builtin(__builtin_amdgcn_cvt_pk_bf16_f32)
__device__ __forceinline__ unsigned int f2bf_pk(float a, float b) {
    union { bf16x2 v; unsigned int u; } cv;
    cv.v = __builtin_amdgcn_cvt_pk_bf16_f32(a, b);
    return cv.u;
}
#else
__device__ __forceinline__ unsigned int f2bf_pk(float a, float b) {
    union { bf16x2 v; unsigned int u; } cv;
    cv.v[0] = (__bf16)a;
    cv.v[1] = (__bf16)b;
    return cv.u;
}
#endif

// ---------------------------------------------------------------------------
// Async global -> LDS 16B copy (CDNA5 GLOBAL_LOAD_ASYNC_TO_LDS_B128, ASYNCcnt)
// with synchronous fallback when the builtin is unavailable.
// ---------------------------------------------------------------------------
#if defined(__AMDGCN__) && __has_builtin(__builtin_amdgcn_global_load_async_to_lds_b128)
#define HAVE_ASYNC_LDS 1
#else
#define HAVE_ASYNC_LDS 0
#endif

__device__ __forceinline__ void copy16_g2l(const unsigned short* g, unsigned short* l) {
#if HAVE_ASYNC_LDS
    __builtin_amdgcn_global_load_async_to_lds_b128(
        (i32x4_g*)g, (i32x4_l*)l, 0, 0);
#else
    *reinterpret_cast<u32x4*>(l) = *reinterpret_cast<const u32x4*>(g);
#endif
}

#if HAVE_ASYNC_LDS
#if __has_builtin(__builtin_amdgcn_s_wait_asynccnt)
#define WAIT_ASYNC() __builtin_amdgcn_s_wait_asynccnt(0)
#else
#define WAIT_ASYNC() asm volatile("s_wait_asynccnt 0" ::: "memory")
#endif
#else
#define WAIT_ASYNC() ((void)0)
#endif

// xor-swizzle add across lanes (group-of-32 ds_swizzle, and_mask=31, xor=m)
#define SWZ_XOR_ADD(x, m) \
    ((x) + __int_as_float(__builtin_amdgcn_ds_swizzle(__float_as_int(x), ((m) << 10) | 0x1f)))

// ---------------------------------------------------------------------------
// Transpose + f32->bf16 convert: src [R][C] f32 -> dst [Cpad][R] bf16 (zero pad)
// grid (R/32, Cpad/32), block (32, 8). R multiple of 32.
// ---------------------------------------------------------------------------
__global__ void transpose_cvt_bf16(const float* __restrict__ src,
                                   unsigned short* __restrict__ dst,
                                   int R, int C, int Cpad) {
    __shared__ float tile[32][33];
    const int r0 = blockIdx.x * 32;
    const int c0 = blockIdx.y * 32;
    const int tx = threadIdx.x, ty = threadIdx.y;
#pragma unroll
    for (int j = 0; j < 4; ++j) {
        int r = r0 + ty + j * 8;
        int c = c0 + tx;
        tile[ty + j * 8][tx] = (c < C) ? src[(size_t)r * C + c] : 0.0f;
    }
    __syncthreads();
#pragma unroll
    for (int j = 0; j < 4; ++j) {
        int c = c0 + ty + j * 8;
        int r = r0 + tx;
        if (c < Cpad)
            dst[(size_t)c * R + r] = f2bf(tile[tx][ty + j * 8]);
    }
}

// ---------------------------------------------------------------------------
// Flat f32 -> bf16 convert, 4 elements/thread.
// ---------------------------------------------------------------------------
__global__ void cvt_bf16(const float* __restrict__ src,
                         unsigned short* __restrict__ dst, int n4) {
    int i = blockIdx.x * blockDim.x + threadIdx.x;
    if (i < n4) {
        f32x4 v = reinterpret_cast<const f32x4*>(src)[i];
        u32x2 o;
        o[0] = f2bf_pk(v[0], v[1]);
        o[1] = f2bf_pk(v[2], v[3]);
        reinterpret_cast<u32x2*>(dst)[i] = o;
    }
}

// ---------------------------------------------------------------------------
// Projection GEMM: H[b][n] = bf16( X[b][:] . WT[n][:] + bias[n] )
// X f32 [8192][2048], WT bf16 [1600][2048], H bf16 [8192][1600]
// Block tile 128x160, BK=32, 256 threads (8 waves: 4M x 2N),
// wave tile = 2 Mtiles x 5 Ntiles of 16x16x32 bf16 WMMA.
// grid (1600/160=10, 8192/128=64)
// ---------------------------------------------------------------------------
__global__ __launch_bounds__(256) void proj_gemm(
    const float* __restrict__ X, const unsigned short* __restrict__ WT,
    const float* __restrict__ bias, unsigned short* __restrict__ H) {
    __shared__ unsigned short As[128 * 40];   // stride 40 ushorts = 80B (5x u32x4)
    __shared__ unsigned short Bs[160 * 40];

    const int tid  = threadIdx.x;
    const int lane = tid & 31;
    const int wave = tid >> 5;
    const int lo   = lane & 15, hi = lane >> 4;
    const int wm   = wave & 3;    // M strip (32 rows)
    const int wn   = wave >> 2;   // N strip (80 cols)
    const int bm   = blockIdx.y * 128;
    const int n0   = blockIdx.x * 160;

    floatx8 acc[2][5];
#pragma unroll
    for (int i = 0; i < 2; ++i)
#pragma unroll
        for (int j = 0; j < 5; ++j) acc[i][j] = (floatx8)0.0f;

    for (int k0 = 0; k0 < DIN; k0 += 32) {
        // stage A: 128x32 f32 -> bf16 in LDS (packed conversion)
#pragma unroll
        for (int j = 0; j < 4; ++j) {
            int lin = tid + j * 256;                 // 0..1023
            int r = lin >> 3, c4 = (lin & 7) * 4;
            f32x4 v = *reinterpret_cast<const f32x4*>(
                &X[(size_t)(bm + r) * DIN + k0 + c4]);
            u32x2 o;
            o[0] = f2bf_pk(v[0], v[1]);
            o[1] = f2bf_pk(v[2], v[3]);
            *reinterpret_cast<u32x2*>(&As[r * 40 + c4]) = o;
        }
        // stage B: 160x32 bf16 copy (640 x 16B) -- async to LDS when available
#pragma unroll
        for (int j = 0; j < 3; ++j) {
            int lin = tid + j * 256;
            if (lin < 640) {
                int r = lin >> 2, c8 = (lin & 3) * 8;
                copy16_g2l(&WT[(size_t)(n0 + r) * DIN + k0 + c8], &Bs[r * 40 + c8]);
            }
        }
        if (k0 + 32 < DIN)
            __builtin_prefetch(&WT[(size_t)(n0 + (tid >> 2)) * DIN + k0 + 32], 0, 3);
        WAIT_ASYNC();
        __syncthreads();

        AFrag a[2], b[5];
#pragma unroll
        for (int mt = 0; mt < 2; ++mt) {
            int row = wm * 32 + mt * 16 + lo;
            const u32x4* pr = reinterpret_cast<const u32x4*>(&As[row * 40]);
            a[mt].q[0] = pr[hi];       // K = hi*8 + 0..7
            a[mt].q[1] = pr[2 + hi];   // K = 16 + hi*8 + 0..7
        }
#pragma unroll
        for (int nt = 0; nt < 5; ++nt) {
            int row = wn * 80 + nt * 16 + lo;
            const u32x4* pr = reinterpret_cast<const u32x4*>(&Bs[row * 40]);
            b[nt].q[0] = pr[hi];
            b[nt].q[1] = pr[2 + hi];
        }
#pragma unroll
        for (int mt = 0; mt < 2; ++mt)
#pragma unroll
            for (int nt = 0; nt < 5; ++nt)
                acc[mt][nt] = __builtin_amdgcn_wmma_f32_16x16x32_bf16(
                    false, a[mt].v, false, b[nt].v, (short)0, acc[mt][nt],
                    false, false);
        __syncthreads();
    }

#pragma unroll
    for (int mt = 0; mt < 2; ++mt)
#pragma unroll
        for (int nt = 0; nt < 5; ++nt) {
            int col = n0 + wn * 80 + nt * 16 + lo;
            float bv = bias[col];
#pragma unroll
            for (int v = 0; v < 8; ++v) {
                int row = bm + wm * 32 + mt * 16 + hi * 8 + v;
                H[(size_t)row * MMDIM + col] = f2bf(acc[mt][nt][v] + bv);
            }
        }
}

// ---------------------------------------------------------------------------
// Fused bilinear: per (chunk c, 128-row tile):
//   g[b, s*80+t] = h0[b,c,s]*h1[b,c,t]  (built in registers)
//   z[b,o] = sum_k g[b,k]*Wb[c,o,k] + bb[c,o]; signed-sqrt; per-row L2 norm
// 8 waves: wave w owns rows w*16..w*16+15, all 5 Ntiles (80 outputs).
// grid (20, 64)
// ---------------------------------------------------------------------------
__global__ __launch_bounds__(256) void bilinear_kernel(
    const unsigned short* __restrict__ H0, const unsigned short* __restrict__ H1,
    const unsigned short* __restrict__ WB, const float* __restrict__ BB,
    unsigned short* __restrict__ Z) {
    __shared__ unsigned short h0s[128 * 88];   // stride 88 ushorts = 176B
    __shared__ unsigned short h1s[128 * 88];

    const int tid  = threadIdx.x;
    const int lane = tid & 31;
    const int wave = tid >> 5;
    const int lo   = lane & 15, hi = lane >> 4;
    const int c    = blockIdx.x;
    const int bm   = blockIdx.y * 128;

    // stage h0/h1 chunk tiles: 128 x 80 bf16 each (1280 x 16B per tensor)
#pragma unroll
    for (int j = 0; j < 5; ++j) {
        int lin = tid + j * 256;             // 0..1279
        int r = lin / 10, c8 = (lin % 10) * 8;
        size_t g = (size_t)(bm + r) * MMDIM + c * CSZ + c8;
        copy16_g2l(&H0[g], &h0s[r * 88 + c8]);
        copy16_g2l(&H1[g], &h1s[r * 88 + c8]);
    }
    WAIT_ASYNC();
    __syncthreads();

    const int row = wave * 16 + lo;   // A-matrix row this lane feeds
    const unsigned short* h0r = &h0s[row * 88];
    const unsigned short* h1r = &h1s[row * 88];
    const unsigned short* wbc = WB + (size_t)c * CSZ * KBIL;

    floatx8 acc[5];
#pragma unroll
    for (int j = 0; j < 5; ++j) acc[j] = (floatx8)0.0f;

    for (int k0 = 0; k0 < KBIL; k0 += 32) {
        // build A fragment on the fly: g[row, k] in bf16 (packed cvt)
        AFrag a;
#pragma unroll
        for (int p = 0; p < 8; ++p) {
            int i = p * 2;
            int ka = k0 + ((i < 8) ? (hi * 8 + i) : (8 + hi * 8 + i));
            int sa = ka / CSZ, ta = ka - sa * CSZ;
            int kb = ka + 1;
            int sb = kb / CSZ, tb = kb - sb * CSZ;
            float pa = bf2f(h0r[sa]) * bf2f(h1r[ta]);
            float pb = bf2f(h0r[sb]) * bf2f(h1r[tb]);
            a.w[p] = f2bf_pk(pa, pb);
        }
        AFrag b[5];
#pragma unroll
        for (int nt = 0; nt < 5; ++nt) {
            const u32x4* pr = reinterpret_cast<const u32x4*>(
                wbc + (size_t)(nt * 16 + lo) * KBIL + k0);
            b[nt].q[0] = pr[hi];
            b[nt].q[1] = pr[2 + hi];
        }
        if (k0 + 32 < KBIL)
            __builtin_prefetch(wbc + (size_t)lo * KBIL + k0 + 32, 0, 3);
#pragma unroll
        for (int nt = 0; nt < 5; ++nt)
            acc[nt] = __builtin_amdgcn_wmma_f32_16x16x32_bf16(
                false, a.v, false, b[nt].v, (short)0, acc[nt], false, false);
    }

    // epilogue: bias + signed sqrt + per-row L2 norm over the 80 outputs
    float ss[8];
#pragma unroll
    for (int v = 0; v < 8; ++v) ss[v] = 0.0f;
#pragma unroll
    for (int nt = 0; nt < 5; ++nt) {
        float bv = BB[c * CSZ + nt * 16 + lo];
#pragma unroll
        for (int v = 0; v < 8; ++v) {
            float z = acc[nt][v] + bv;
            float s = sqrtf(fabsf(z));
            z = copysignf(s, z);
            acc[nt][v] = z;
            ss[v] += z * z;
        }
    }
#pragma unroll
    for (int v = 0; v < 8; ++v) {
        float x = ss[v];
        x = SWZ_XOR_ADD(x, 1);
        x = SWZ_XOR_ADD(x, 2);
        x = SWZ_XOR_ADD(x, 4);
        x = SWZ_XOR_ADD(x, 8);
        float nrm = sqrtf(x);
        ss[v] = 1.0f / fmaxf(nrm, 1e-12f);
    }
#pragma unroll
    for (int nt = 0; nt < 5; ++nt)
#pragma unroll
        for (int v = 0; v < 8; ++v) {
            int r = bm + wave * 16 + hi * 8 + v;
            int o = nt * 16 + lo;
            Z[(size_t)r * MMDIM + c * CSZ + o] = f2bf(acc[nt][v] * ss[v]);
        }
}

// ---------------------------------------------------------------------------
// Output GEMM: OUT[b][n] = Z[b][:] . WoutT[n][:] + bout[n]  (f32 out)
// Z bf16 [8192][1600], WoutT bf16 [3040][1600] (zero padded rows >= 3000)
// grid (3040/160=19, 64)
// ---------------------------------------------------------------------------
__global__ __launch_bounds__(256) void out_gemm(
    const unsigned short* __restrict__ Zb, const unsigned short* __restrict__ WT,
    const float* __restrict__ bias, float* __restrict__ OUT) {
    __shared__ unsigned short As[128 * 40];
    __shared__ unsigned short Bs[160 * 40];

    const int tid  = threadIdx.x;
    const int lane = tid & 31;
    const int wave = tid >> 5;
    const int lo   = lane & 15, hi = lane >> 4;
    const int wm   = wave & 3;
    const int wn   = wave >> 2;
    const int bm   = blockIdx.y * 128;
    const int n0   = blockIdx.x * 160;

    floatx8 acc[2][5];
#pragma unroll
    for (int i = 0; i < 2; ++i)
#pragma unroll
        for (int j = 0; j < 5; ++j) acc[i][j] = (floatx8)0.0f;

    for (int k0 = 0; k0 < MMDIM; k0 += 32) {
        // stage A: 128x32 bf16 copy (512 x 16B) -- async
#pragma unroll
        for (int j = 0; j < 2; ++j) {
            int lin = tid + j * 256;
            int r = lin >> 2, c8 = (lin & 3) * 8;
            copy16_g2l(&Zb[(size_t)(bm + r) * MMDIM + k0 + c8], &As[r * 40 + c8]);
        }
        // stage B: 160x32 bf16 copy (640 x 16B) -- async
#pragma unroll
        for (int j = 0; j < 3; ++j) {
            int lin = tid + j * 256;
            if (lin < 640) {
                int r = lin >> 2, c8 = (lin & 3) * 8;
                copy16_g2l(&WT[(size_t)(n0 + r) * MMDIM + k0 + c8], &Bs[r * 40 + c8]);
            }
        }
        WAIT_ASYNC();
        __syncthreads();

        AFrag a[2], b[5];
#pragma unroll
        for (int mt = 0; mt < 2; ++mt) {
            int row = wm * 32 + mt * 16 + lo;
            const u32x4* pr = reinterpret_cast<const u32x4*>(&As[row * 40]);
            a[mt].q[0] = pr[hi];
            a[mt].q[1] = pr[2 + hi];
        }
#pragma unroll
        for (int nt = 0; nt < 5; ++nt) {
            int row = wn * 80 + nt * 16 + lo;
            const u32x4* pr = reinterpret_cast<const u32x4*>(&Bs[row * 40]);
            b[nt].q[0] = pr[hi];
            b[nt].q[1] = pr[2 + hi];
        }
#pragma unroll
        for (int mt = 0; mt < 2; ++mt)
#pragma unroll
            for (int nt = 0; nt < 5; ++nt)
                acc[mt][nt] = __builtin_amdgcn_wmma_f32_16x16x32_bf16(
                    false, a[mt].v, false, b[nt].v, (short)0, acc[mt][nt],
                    false, false);
        __syncthreads();
    }

#pragma unroll
    for (int mt = 0; mt < 2; ++mt)
#pragma unroll
        for (int nt = 0; nt < 5; ++nt) {
            int col = n0 + wn * 80 + nt * 16 + lo;
            if (col < DOUT) {
                float bv = bias[col];
#pragma unroll
                for (int v = 0; v < 8; ++v) {
                    int row = bm + wm * 32 + mt * 16 + hi * 8 + v;
                    OUT[(size_t)row * DOUT + col] = acc[mt][nt][v] + bv;
                }
            }
        }
}

// ---------------------------------------------------------------------------
extern "C" void kernel_launch(void* const* d_in, const int* in_sizes, int n_in,
                              void* d_out, int out_size, void* d_ws, size_t ws_size,
                              hipStream_t stream) {
    const float* x0   = (const float*)d_in[0];
    const float* x1   = (const float*)d_in[1];
    const float* W0   = (const float*)d_in[2];
    const float* b0   = (const float*)d_in[3];
    const float* W1   = (const float*)d_in[4];
    const float* b1   = (const float*)d_in[5];
    const float* Wb   = (const float*)d_in[6];
    const float* bb   = (const float*)d_in[7];
    const float* Wout = (const float*)d_in[8];
    const float* bout = (const float*)d_in[9];
    float* out = (float*)d_out;

    // workspace layout (all bf16 as ushort), each region multiple of 256 B
    size_t off = 0;
    auto alloc = [&](size_t bytes) {
        void* p = (char*)d_ws + off;
        off += bytes;
        return p;
    };
    unsigned short* W0T   = (unsigned short*)alloc((size_t)MMDIM * DIN * 2);
    unsigned short* W1T   = (unsigned short*)alloc((size_t)MMDIM * DIN * 2);
    unsigned short* WbB   = (unsigned short*)alloc((size_t)CHUNKS * CSZ * KBIL * 2);
    unsigned short* WoutT = (unsigned short*)alloc((size_t)NPAD * MMDIM * 2);
    unsigned short* h0B   = (unsigned short*)alloc((size_t)B_ROWS * MMDIM * 2);
    unsigned short* h1B   = (unsigned short*)alloc((size_t)B_ROWS * MMDIM * 2);
    unsigned short* zB    = (unsigned short*)alloc((size_t)B_ROWS * MMDIM * 2);
    (void)ws_size; (void)in_sizes; (void)n_in; (void)out_size;

    dim3 tb(32, 8);
    // W0^T, W1^T : [2048][1600] -> [1600][2048]
    transpose_cvt_bf16<<<dim3(DIN / 32, MMDIM / 32), tb, 0, stream>>>(W0, W0T, DIN, MMDIM, MMDIM);
    transpose_cvt_bf16<<<dim3(DIN / 32, MMDIM / 32), tb, 0, stream>>>(W1, W1T, DIN, MMDIM, MMDIM);
    // Wout^T : [1600][3000] -> [3040][1600], zero padded
    transpose_cvt_bf16<<<dim3(MMDIM / 32, NPAD / 32), tb, 0, stream>>>(Wout, WoutT, MMDIM, DOUT, NPAD);
    // Wb flat convert: 20*80*6400 = 10,240,000 elems
    cvt_bf16<<<(CHUNKS * CSZ * KBIL / 4 + 255) / 256, 256, 0, stream>>>(
        Wb, WbB, CHUNKS * CSZ * KBIL / 4);

    proj_gemm<<<dim3(MMDIM / 160, B_ROWS / 128), 256, 0, stream>>>(x0, W0T, b0, h0B);
    proj_gemm<<<dim3(MMDIM / 160, B_ROWS / 128), 256, 0, stream>>>(x1, W1T, b1, h1B);

    bilinear_kernel<<<dim3(CHUNKS, B_ROWS / 128), 256, 0, stream>>>(h0B, h1B, WbB, bb, zB);

    out_gemm<<<dim3(NPAD / 160, B_ROWS / 128), 256, 0, stream>>>(zB, WoutT, bout, out);
}
